// CausalAttention_11458972745858
// MI455X (gfx1250) — matrix operand order repaired and checked by reading, not compiled, not run on hardware
//
#include <hip/hip_runtime.h>
#include <stdint.h>

// ---------------- types ----------------
typedef __bf16 bf16_t;
typedef __attribute__((ext_vector_type(16))) __bf16 v16bf;
typedef __attribute__((ext_vector_type(8)))  float  v8f;

// ---------------- helpers ----------------
__device__ __forceinline__ uint16_t f2bf(float f) {
    uint32_t u = __float_as_uint(f);
    uint32_t r = u + 0x7FFFu + ((u >> 16) & 1u);   // round-to-nearest-even
    return (uint16_t)(r >> 16);
}

// A-fragment: 16x32 bf16, row r = lane&15, per-half K pattern {0..7,16..23}/{8..15,24..31}
__device__ __forceinline__ v16bf load_a_frag(const uint16_t* base, int ld) {
    const int lane = threadIdx.x & 31;
    const int r = lane & 15, h = lane >> 4;
    const uint16_t* p = base + (size_t)r * ld + h * 8;
    union { v16bf v; uint32_t u[8]; } f;
#pragma unroll
    for (int q = 0; q < 8; ++q) {
        int off = ((q < 4) ? 0 : 16) + (q & 3) * 2;
        f.u[q] = *reinterpret_cast<const uint32_t*>(p + off);
    }
    return f.v;
}

// B-fragment: 32x16 bf16 read from B^T row-major (lane = N row, per-half K pattern {0..7,16..23}/{8..15,24..31})
__device__ __forceinline__ v16bf load_b_frag(const uint16_t* base, int ld) {
    const int lane = threadIdx.x & 31;
    const int n = lane & 15, h = lane >> 4;
    const uint16_t* p = base + (size_t)n * ld + h * 8;
    union { v16bf v; uint32_t u[8]; } f;
#pragma unroll
    for (int q = 0; q < 8; ++q)
        f.u[q] = *reinterpret_cast<const uint32_t*>(p + 2 * q + ((q < 4) ? 0 : 8));
    return f.v;
}

__device__ __forceinline__ v8f wmma_bf16(v16bf a, v16bf b, v8f c) {
    return __builtin_amdgcn_wmma_f32_16x16x32_bf16(false, a, false, b, (short)0, c, false, false);
}

// ---------------- fp32 -> bf16 pack ----------------
__global__ __launch_bounds__(256) void k_f32_to_bf16(const float* __restrict__ in,
                                                     uint16_t* __restrict__ out, int n4) {
    int i = blockIdx.x * blockDim.x + threadIdx.x;
    if (i < n4) {
        float4 v = reinterpret_cast<const float4*>(in)[i];
        uint2 o;
        o.x = (uint32_t)f2bf(v.x) | ((uint32_t)f2bf(v.y) << 16);
        o.y = (uint32_t)f2bf(v.z) | ((uint32_t)f2bf(v.w) << 16);
        reinterpret_cast<uint2*>(out)[i] = o;
    }
}

// ---------------- QKV projection: [4096,1024] x [3072,1024]^T ----------------
// One wave computes a 64x64 tile; writes Q[B,H,T,Dk], K[B,H,T,Dk], Vt[B,H,Dk,T] (bf16)
__global__ __launch_bounds__(32) void k_qkv_gemm(const uint16_t* __restrict__ X,
                                                 const uint16_t* __restrict__ W,
                                                 uint16_t* __restrict__ Qo,
                                                 uint16_t* __restrict__ Ko,
                                                 uint16_t* __restrict__ Vt) {
    const int lane = threadIdx.x;
    const int nlo = lane & 15, hl = lane >> 4;
    const int m0 = blockIdx.y * 64;
    const int n0 = blockIdx.x * 64;
    v8f acc[4][4] = {};
    for (int kk = 0; kk < 1024; kk += 32) {
        v16bf bf[4];
#pragma unroll
        for (int j = 0; j < 4; ++j)
            bf[j] = load_b_frag(W + (size_t)(n0 + j * 16) * 1024 + kk, 1024);
#pragma unroll
        for (int i = 0; i < 4; ++i) {
            v16bf af = load_a_frag(X + (size_t)(m0 + i * 16) * 1024 + kk, 1024);
#pragma unroll
            for (int j = 0; j < 4; ++j)
                acc[i][j] = wmma_bf16(af, bf[j], acc[i][j]);
        }
    }
#pragma unroll
    for (int j = 0; j < 4; ++j) {
        const int n = n0 + j * 16 + nlo;
        const int sel = n >> 10, f = n & 1023, hh = f >> 6, dk = f & 63;
#pragma unroll
        for (int i = 0; i < 4; ++i) {
#pragma unroll
            for (int vi = 0; vi < 8; ++vi) {
                const int m = m0 + i * 16 + vi + 8 * hl;
                const int b = m >> 11, t = m & 2047;
                const uint16_t val = f2bf(acc[i][j][vi]);
                if (sel == 0)      Qo[(((size_t)(b * 16 + hh) * 2048 + t) << 6) + dk] = val;
                else if (sel == 1) Ko[(((size_t)(b * 16 + hh) * 2048 + t) << 6) + dk] = val;
                else               Vt[(((size_t)(b * 16 + hh) * 64 + dk) << 11) + t] = val;
            }
        }
    }
}

// ---------------- flash attention (causal, online softmax) ----------------
// 8 waves/block; each wave owns a 16-row query block of one (batch,head).
__global__ __launch_bounds__(256) void k_flash(const uint16_t* __restrict__ Q,
                                               const uint16_t* __restrict__ K,
                                               const uint16_t* __restrict__ Vt,
                                               uint16_t* __restrict__ ctx) {
    __shared__ uint16_t plds[8 * 16 * 32];   // per-wave 16x32 bf16 P tile
    const int lane = threadIdx.x & 31;
    const int widx = threadIdx.x >> 5;
    const int nlo = lane & 15, hl = lane >> 4;
    const int w = blockIdx.x * 8 + widx;     // 0..4095
    const int bh = w & 31;                   // interleave heads for load balance
    const int qb = w >> 5;
    const int b = bh >> 4, hh = bh & 15;
    const uint16_t* Qh = Q  + (size_t)bh * 2048 * 64;
    const uint16_t* Kh = K  + (size_t)bh * 2048 * 64;
    const uint16_t* Vh = Vt + (size_t)bh * 64 * 2048;
    const int qbase = qb * 16;
    uint16_t* myp = plds + widx * 512;

    const v16bf qa0 = load_a_frag(Qh + (size_t)qbase * 64, 64);
    const v16bf qa1 = load_a_frag(Qh + (size_t)qbase * 64 + 32, 64);

    v8f acc[4] = {};
    float mr[8], lr[8];
#pragma unroll
    for (int vi = 0; vi < 8; ++vi) { mr[vi] = -1e30f; lr[vi] = 0.0f; }

    const int nt = (qbase + 16 + 31) >> 5;   // 32-key tiles needed (causal)
    for (int kt = 0; kt < nt; ++kt) {
        const int k0 = kt * 32;
        // scores: S = Q(16x64) * K^T(64x32), split dk in two WMMAs per 16-key group
        v8f s0 = {}, s1 = {};
        s0 = wmma_bf16(qa0, load_b_frag(Kh + (size_t)k0 * 64, 64), s0);
        s0 = wmma_bf16(qa1, load_b_frag(Kh + (size_t)k0 * 64 + 32, 64), s0);
        s1 = wmma_bf16(qa0, load_b_frag(Kh + (size_t)(k0 + 16) * 64, 64), s1);
        s1 = wmma_bf16(qa1, load_b_frag(Kh + (size_t)(k0 + 16) * 64 + 32, 64), s1);

        float p0[8], p1[8], sc[8];
#pragma unroll
        for (int vi = 0; vi < 8; ++vi) {
            const int qrow = qbase + vi + 8 * hl;
            float a = s0[vi] * 0.125f;                 // 1/sqrt(64)
            float c = s1[vi] * 0.125f;
            if (k0 + nlo > qrow)      a = -1e30f;      // causal mask
            if (k0 + 16 + nlo > qrow) c = -1e30f;
            float rm = fmaxf(a, c);                     // row max across 16 lanes
            rm = fmaxf(rm, __shfl_xor(rm, 1));
            rm = fmaxf(rm, __shfl_xor(rm, 2));
            rm = fmaxf(rm, __shfl_xor(rm, 4));
            rm = fmaxf(rm, __shfl_xor(rm, 8));
            const float mn = fmaxf(mr[vi], rm);
            sc[vi] = __expf(mr[vi] - mn);
            const float e0 = __expf(a - mn);
            const float e1 = __expf(c - mn);
            p0[vi] = e0; p1[vi] = e1;
            float rs = e0 + e1;                         // row sum across 16 lanes
            rs += __shfl_xor(rs, 1);
            rs += __shfl_xor(rs, 2);
            rs += __shfl_xor(rs, 4);
            rs += __shfl_xor(rs, 8);
            lr[vi] = lr[vi] * sc[vi] + rs;
            mr[vi] = mn;
        }
#pragma unroll
        for (int j = 0; j < 4; ++j)
#pragma unroll
            for (int vi = 0; vi < 8; ++vi)
                acc[j][vi] *= sc[vi];
        // transpose P: C-layout -> A-layout via per-wave LDS slice (in-order DS)
#pragma unroll
        for (int vi = 0; vi < 8; ++vi) {
            const int row = vi + 8 * hl;
            myp[row * 32 + nlo]      = f2bf(p0[vi]);
            myp[row * 32 + 16 + nlo] = f2bf(p1[vi]);
        }
        const v16bf pa = load_a_frag(myp, 32);
        // acc += P(16x32) * V(32x64), B-fragments contiguous thanks to V^T layout
#pragma unroll
        for (int j = 0; j < 4; ++j)
            acc[j] = wmma_bf16(pa, load_b_frag(Vh + (size_t)(j * 16) * 2048 + k0, 2048), acc[j]);
    }

    float inv[8];
#pragma unroll
    for (int vi = 0; vi < 8; ++vi) inv[vi] = 1.0f / lr[vi];
#pragma unroll
    for (int j = 0; j < 4; ++j)
#pragma unroll
        for (int vi = 0; vi < 8; ++vi) {
            const int t = qbase + vi + 8 * hl;
            ctx[(size_t)(b * 2048 + t) * 1024 + hh * 64 + j * 16 + nlo] =
                f2bf(acc[j][vi] * inv[vi]);
        }
}

// ---------------- output projection: [4096,1024] x [1024,1024]^T -> f32 ----------------
__global__ __launch_bounds__(32) void k_out_proj(const uint16_t* __restrict__ Ctx,
                                                 const uint16_t* __restrict__ Wo,
                                                 float* __restrict__ out) {
    const int lane = threadIdx.x;
    const int nlo = lane & 15, hl = lane >> 4;
    const int m0 = blockIdx.y * 64;
    const int n0 = blockIdx.x * 64;
    v8f acc[4][4] = {};
    for (int kk = 0; kk < 1024; kk += 32) {
        v16bf bf[4];
#pragma unroll
        for (int j = 0; j < 4; ++j)
            bf[j] = load_b_frag(Wo + (size_t)(n0 + j * 16) * 1024 + kk, 1024);
#pragma unroll
        for (int i = 0; i < 4; ++i) {
            v16bf af = load_a_frag(Ctx + (size_t)(m0 + i * 16) * 1024 + kk, 1024);
#pragma unroll
            for (int j = 0; j < 4; ++j)
                acc[i][j] = wmma_bf16(af, bf[j], acc[i][j]);
        }
    }
#pragma unroll
    for (int i = 0; i < 4; ++i)
#pragma unroll
        for (int j = 0; j < 4; ++j)
#pragma unroll
            for (int vi = 0; vi < 8; ++vi) {
                const int m = m0 + i * 16 + vi + 8 * hl;
                const int n = n0 + j * 16 + nlo;
                out[(size_t)m * 1024 + n] = acc[i][j][vi];
            }
}

// ---------------- launch ----------------
extern "C" void kernel_launch(void* const* d_in, const int* in_sizes, int n_in,
                              void* d_out, int out_size, void* d_ws, size_t ws_size,
                              hipStream_t stream) {
    (void)in_sizes; (void)n_in; (void)out_size; (void)ws_size;
    const float* x    = (const float*)d_in[0];   // [2,2048,1024]
    const float* wqkv = (const float*)d_in[1];   // [3072,1024]
    const float* wo   = (const float*)d_in[2];   // [1024,1024]

    uint8_t* ws = (uint8_t*)d_ws;
    uint16_t* xb    = (uint16_t*)(ws);                          // 8 MB
    uint16_t* wqkvb = (uint16_t*)(ws + ((size_t)8  << 20));     // 6 MB
    uint16_t* wob   = (uint16_t*)(ws + ((size_t)14 << 20));     // 2 MB
    uint16_t* Qb    = (uint16_t*)(ws + ((size_t)16 << 20));     // 8 MB
    uint16_t* Kb    = (uint16_t*)(ws + ((size_t)24 << 20));     // 8 MB
    uint16_t* Vtb   = (uint16_t*)(ws + ((size_t)32 << 20));     // 8 MB
    uint16_t* ctxb  = xb;   // x_bf16 dead after QKV GEMM; reuse for context

    k_f32_to_bf16<<<4096, 256, 0, stream>>>(x,    xb,    1048576); // 4,194,304/4
    k_f32_to_bf16<<<3072, 256, 0, stream>>>(wqkv, wqkvb,  786432);
    k_f32_to_bf16<<<1024, 256, 0, stream>>>(wo,   wob,    262144);

    k_qkv_gemm<<<dim3(48, 64), 32, 0, stream>>>(xb, wqkvb, Qb, Kb, Vtb);
    k_flash<<<512, 256, 0, stream>>>(Qb, Kb, Vtb, ctxb);
    k_out_proj<<<dim3(16, 64), 32, 0, stream>>>(ctxb, wob, (float*)d_out);
}
